// TemporalTFB_5703716569216
// MI455X (gfx1250) — matrix operand order repaired and checked by reading, not compiled, not run on hardware
//
#include <hip/hip_runtime.h>
#include <math.h>

// ---------------- problem constants ----------------
namespace {
constexpr int kB    = 4;
constexpr int kS    = 2048;
constexpr int kM    = 1024;
constexpr int kH    = 16;
constexpr int kDH   = 64;
constexpr int kHID  = 4096;
constexpr int kRows = kB * kS;   // 8192
constexpr int kN3   = 3 * kM;    // 3072
}

typedef __attribute__((ext_vector_type(16))) __bf16        v16bf;
typedef __attribute__((ext_vector_type(8)))  float         v8f;
typedef __attribute__((ext_vector_type(8)))  unsigned int  v8u;
typedef __attribute__((ext_vector_type(4)))  unsigned int  u32x4;
typedef __attribute__((ext_vector_type(8)))  int           i32x8;
typedef __attribute__((ext_vector_type(4)))  int           i32x4;

#if defined(__HIP_DEVICE_COMPILE__) && __has_builtin(__builtin_amdgcn_tensor_load_to_lds)
#define USE_TDM 1
#else
#define USE_TDM 0
#endif

__device__ __forceinline__ unsigned short f32_to_bf16(float f) {
  unsigned int u = __float_as_uint(f);
  u += 0x7FFFu + ((u >> 16) & 1u);   // round-to-nearest-even
  return (unsigned short)(u >> 16);
}

// ---------------- WMMA fragment loaders (ISA 7.12.2, wave32) ----------------
// Work from global or LDS pointers (addrspace inferred -> ds_load_b128 for LDS).
// A: 16x32 bf16 row-major, row stride lda.
__device__ __forceinline__ v16bf load_a_frag(const unsigned short* A, int lda) {
  int lane = (int)(threadIdx.x & 31u);
  int row = lane & 15, g = lane >> 4;
  const uint4* p0 = (const uint4*)(A + (size_t)row * lda + g * 16);
  const uint4* p1 = (const uint4*)(A + (size_t)row * lda + g * 16 + 8);
  uint4 lo = *p0, hi = *p1;
  v8u u = {lo.x, lo.y, lo.z, lo.w, hi.x, hi.y, hi.z, hi.w};
  return __builtin_bit_cast(v16bf, u);
}

// B: 32x16 where B(k,n) = W[n][k]; W row-major [N,K], row stride ldw.
__device__ __forceinline__ v16bf load_b_frag_wT(const unsigned short* W, int ldw) {
  int lane = (int)(threadIdx.x & 31u);
  int n = lane & 15, g = lane >> 4;
  const uint4* p0 = (const uint4*)(W + (size_t)n * ldw + g * 16);
  const uint4* p1 = (const uint4*)(W + (size_t)n * ldw + g * 16 + 8);
  uint4 lo = *p0, hi = *p1;
  v8u u = {lo.x, lo.y, lo.z, lo.w, hi.x, hi.y, hi.z, hi.w};
  return __builtin_bit_cast(v16bf, u);
}

#define WMMA_BF16(a, b, c) \
  __builtin_amdgcn_wmma_f32_16x16x32_bf16(false, (a), false, (b), (short)0, (c), false, false)

// ---------------- TDM: 2-D tile DMA  global -> LDS  (ISA ch.8 D# layout) ----------------
#if USE_TDM
__device__ __forceinline__ void tdm_load_2d(unsigned lds_off, const void* gaddr,
                                            unsigned tile_x, unsigned tile_y,
                                            unsigned stride_elems,
                                            unsigned tensor_x, unsigned tensor_y) {
  unsigned long long ga = (unsigned long long)(size_t)gaddr;
  u32x4 g0;
  g0[0] = 1u;                                        // count=1, user desc
  g0[1] = lds_off;                                   // lds_addr (bytes)
  g0[2] = (unsigned)(ga & 0xFFFFFFFFu);              // global_addr[31:0]
  g0[3] = (unsigned)((ga >> 32) & 0x01FFFFFFu)       // global_addr[56:32]
          | (2u << 30);                              // type = 2 ("image")
  i32x8 g1;
  g1[0] = (int)(1u << 16);                           // workgroup_mask=0, data_size=1 (2B)
  g1[1] = (int)((tensor_x & 0xFFFFu) << 16);         // atomic_addr=0 | tensor_dim0[15:0]
  g1[2] = (int)(((tensor_x >> 16) & 0xFFFFu) |       // tensor_dim0[31:16]
                ((tensor_y & 0xFFFFu) << 16));       // tensor_dim1[15:0]
  g1[3] = (int)(((tensor_y >> 16) & 0xFFFFu) |       // tensor_dim1[31:16]
                ((tile_x & 0xFFFFu) << 16));         // tile_dim0
  g1[4] = (int)(tile_y & 0xFFFFu);                   // tile_dim1 | tile_dim2=0
  g1[5] = (int)stride_elems;                         // tensor_dim0_stride[31:0]
  g1[6] = 0;                                         // stride0[47:32] | stride1[15:0]
  g1[7] = 0;                                         // tensor_dim1_stride rest
  i32x4 z4 = {0, 0, 0, 0};                           // D# groups 2/3 unused (2-D tile)
  i32x8 z8 = {0, 0, 0, 0, 0, 0, 0, 0};
  __builtin_amdgcn_tensor_load_to_lds(g0, g1, z4, z4, z8, 0);
}
#endif

// ---------------- elementwise fp32 -> bf16 ----------------
__global__ __launch_bounds__(256)
void cvt_f32_bf16(const float* __restrict__ src, unsigned short* __restrict__ dst, int n) {
  int i = blockIdx.x * blockDim.x + threadIdx.x;
  int stride = gridDim.x * blockDim.x;
  for (; i < n; i += stride) dst[i] = f32_to_bf16(src[i]);
}

// ---------------- LayerNorm (fp32 in) -> bf16 out, one row per block ----------------
__global__ __launch_bounds__(256)
void ln_to_bf16(const float* __restrict__ x, const float* __restrict__ gam,
                const float* __restrict__ bet, unsigned short* __restrict__ out) {
  int row = blockIdx.x;
  const float* xr = x + (size_t)row * kM;
  float s = 0.f, s2 = 0.f;
  for (int i = threadIdx.x; i < kM; i += blockDim.x) {
    float v = xr[i]; s += v; s2 += v * v;
  }
  __shared__ float sh1[256], sh2[256];
  sh1[threadIdx.x] = s; sh2[threadIdx.x] = s2;
  __syncthreads();
  for (int off = 128; off > 0; off >>= 1) {
    if ((int)threadIdx.x < off) {
      sh1[threadIdx.x] += sh1[threadIdx.x + off];
      sh2[threadIdx.x] += sh2[threadIdx.x + off];
    }
    __syncthreads();
  }
  float mean = sh1[0] * (1.f / kM);
  float var  = sh2[0] * (1.f / kM) - mean * mean;
  float inv  = rsqrtf(var + 1e-5f);
  unsigned short* orow = out + (size_t)row * kM;
  for (int i = threadIdx.x; i < kM; i += blockDim.x) {
    float v = (xr[i] - mean) * inv * gam[i] + bet[i];
    orow[i] = f32_to_bf16(v);
  }
}

// ---------------- epilogue store for one 16x16 accumulator ----------------
// EPI 0: bf16 | 1: +resid -> f32 | 2: +bias, exact GELU -> bf16 | 3: +bias +resid -> f32
template <int EPI>
__device__ __forceinline__ void store_tile(v8f acc, int row0, int col0, int N,
                                           const float* __restrict__ bias,
                                           const float* __restrict__ resid,
                                           void* __restrict__ outv) {
  int lane  = (int)(threadIdx.x & 31u);
  int col   = col0 + (lane & 15);
  int rbase = row0 + ((lane >> 4) << 3);
  if (EPI == 0) {
    unsigned short* out = (unsigned short*)outv;
#pragma unroll
    for (int r = 0; r < 8; ++r)
      out[(size_t)(rbase + r) * N + col] = f32_to_bf16(acc[r]);
  } else if (EPI == 1) {
    float* out = (float*)outv;
#pragma unroll
    for (int r = 0; r < 8; ++r) {
      size_t idx = (size_t)(rbase + r) * N + col;
      out[idx] = acc[r] + resid[idx];
    }
  } else if (EPI == 2) {
    unsigned short* out = (unsigned short*)outv;
    float bv = bias[col];
#pragma unroll
    for (int r = 0; r < 8; ++r) {
      float xg  = acc[r] + bv;
      float gel = 0.5f * xg * (1.f + erff(xg * 0.70710678118654752f));
      out[(size_t)(rbase + r) * N + col] = f32_to_bf16(gel);
    }
  } else {
    float* out = (float*)outv;
    float bv = bias[col];
#pragma unroll
    for (int r = 0; r < 8; ++r) {
      size_t idx = (size_t)(rbase + r) * N + col;
      out[idx] = acc[r] + bv + resid[idx];
    }
  }
}

// ---------------- block-tiled  C = A[RxK] @ W[NxK]^T ----------------
// Block = 128(M) x 64(N); 8 waves in 4x2; each wave 32x32 (2x2 WMMA accumulators).
// K staged 64 at a time into LDS, double-buffered, filled by the Tensor Data Mover.
template <int EPI>
__global__ __launch_bounds__(256)
void gemm_xwT(const unsigned short* __restrict__ A,
              const unsigned short* __restrict__ W,
              int K, int N,
              const float* __restrict__ bias,
              const float* __restrict__ resid,
              void* __restrict__ outv) {
  __shared__ __align__(16) unsigned short shA[2][128 * 64];  // 2 x 16 KB
  __shared__ __align__(16) unsigned short shW[2][64 * 64];   // 2 x  8 KB
  const int bx = blockIdx.x;                 // N block
  const int by = blockIdx.y;                 // M block
  const int wv = (int)threadIdx.x >> 5;
  const int waveM = wv >> 1, waveN = wv & 1;
  const int rowBlk = by * 128, colBlk = bx * 64;

#if USE_TDM
  const bool issuer = ((int)threadIdx.x < 32);   // wave 0 drives the TDM
#endif

  auto stage = [&](int buf, int k) {
#if USE_TDM
    tdm_load_2d((unsigned)(size_t)&shA[buf][0],
                A + (size_t)rowBlk * K + k, 64u, 128u, (unsigned)K,
                (unsigned)K, 1u << 20);
    tdm_load_2d((unsigned)(size_t)&shW[buf][0],
                W + (size_t)colBlk * K + k, 64u, 64u, (unsigned)K,
                (unsigned)K, 1u << 20);
#else
    for (int i = threadIdx.x; i < 128 * 8; i += 256) {     // A tile, b128 chunks
      int r = i >> 3, c = (i & 7) * 8;
      *(uint4*)&shA[buf][r * 64 + c] =
          *(const uint4*)&A[(size_t)(rowBlk + r) * K + k + c];
    }
    for (int i = threadIdx.x; i < 64 * 8; i += 256) {      // W tile
      int r = i >> 3, c = (i & 7) * 8;
      *(uint4*)&shW[buf][r * 64 + c] =
          *(const uint4*)&W[(size_t)(colBlk + r) * K + k + c];
    }
#endif
  };

  v8f acc00 = {0.f,0.f,0.f,0.f,0.f,0.f,0.f,0.f};
  v8f acc01 = acc00, acc10 = acc00, acc11 = acc00;

  // prologue: fill buffer 0
#if USE_TDM
  if (issuer) { stage(0, 0); __builtin_amdgcn_s_wait_tensorcnt(0); }
#else
  stage(0, 0);
#endif
  __syncthreads();

  for (int k = 0; k < K; k += 64) {
    int cur = (k >> 6) & 1;
    int nxt = cur ^ 1;
    bool more = (k + 64) < K;
#if USE_TDM
    if (more && issuer) stage(nxt, k + 64);   // DMA next slab while we compute
#else
    if (more) stage(nxt, k + 64);
#endif
    const unsigned short* At = &shA[cur][waveM * 32 * 64];
    const unsigned short* Wt = &shW[cur][waveN * 32 * 64];
#pragma unroll
    for (int ks = 0; ks < 64; ks += 32) {
      v16bf a0 = load_a_frag(At + ks, 64);
      v16bf a1 = load_a_frag(At + 16 * 64 + ks, 64);
      v16bf b0 = load_b_frag_wT(Wt + ks, 64);
      v16bf b1 = load_b_frag_wT(Wt + 16 * 64 + ks, 64);
      acc00 = WMMA_BF16(a0, b0, acc00);
      acc01 = WMMA_BF16(a0, b1, acc01);
      acc10 = WMMA_BF16(a1, b0, acc10);
      acc11 = WMMA_BF16(a1, b1, acc11);
    }
    if (more) {
#if USE_TDM
      if (issuer) __builtin_amdgcn_s_wait_tensorcnt(0);
#endif
      __syncthreads();
    }
  }

  int r0 = rowBlk + waveM * 32, c0 = colBlk + waveN * 32;
  store_tile<EPI>(acc00, r0,      c0,      N, bias, resid, outv);
  store_tile<EPI>(acc01, r0,      c0 + 16, N, bias, resid, outv);
  store_tile<EPI>(acc10, r0 + 16, c0,      N, bias, resid, outv);
  store_tile<EPI>(acc11, r0 + 16, c0 + 16, N, bias, resid, outv);
}

// ---------------- KtV[bh] = (K^T @ V) / sqrt(DH), 64x64 per (b,h) ----------------
// One block per (b,h). K/V slabs (32 x 64) staged through LDS by the TDM,
// double-buffered; waves build K^T fragments from LDS (ds reads, single pass
// over global K/V). 8 waves: wave w -> row tile td2=w>>1, col tiles 2*(w&1)+{0,1}.
__global__ __launch_bounds__(256)
void ktv_kernel(const unsigned short* __restrict__ qkv,
                unsigned short* __restrict__ ktv) {
  __shared__ __align__(16) unsigned short shK[2][32 * 64];  // 2 x 4 KB
  __shared__ __align__(16) unsigned short shV[2][32 * 64];  // 2 x 4 KB
  const int bh = blockIdx.x;
  const int b = bh >> 4, h = bh & 15;
  const unsigned short* Kg = qkv + (size_t)b * kS * kN3 + kM     + h * kDH;
  const unsigned short* Vg = qkv + (size_t)b * kS * kN3 + 2 * kM + h * kDH;

#if USE_TDM
  const bool issuer = ((int)threadIdx.x < 32);
#endif

  auto stage = [&](int buf, int s) {
#if USE_TDM
    tdm_load_2d((unsigned)(size_t)&shK[buf][0], Kg + (size_t)s * kN3,
                64u, 32u, (unsigned)kN3, (unsigned)kN3, 1u << 20);
    tdm_load_2d((unsigned)(size_t)&shV[buf][0], Vg + (size_t)s * kN3,
                64u, 32u, (unsigned)kN3, (unsigned)kN3, 1u << 20);
#else
    int r = (int)threadIdx.x >> 3, c = ((int)threadIdx.x & 7) * 8;  // 32 rows x 8 chunks
    *(uint4*)&shK[buf][r * 64 + c] = *(const uint4*)&Kg[(size_t)(s + r) * kN3 + c];
    *(uint4*)&shV[buf][r * 64 + c] = *(const uint4*)&Vg[(size_t)(s + r) * kN3 + c];
#endif
  };

  const int wv   = (int)threadIdx.x >> 5;
  const int lane = (int)(threadIdx.x & 31u);
  const int td2  = wv >> 1;            // output row tile 0..3
  const int tc0  = (wv & 1) * 2;       // output col tiles tc0, tc0+1
  const int m    = lane & 15, g = lane >> 4;

  v8f acc0 = {0.f,0.f,0.f,0.f,0.f,0.f,0.f,0.f};
  v8f acc1 = acc0;

#if USE_TDM
  if (issuer) { stage(0, 0); __builtin_amdgcn_s_wait_tensorcnt(0); }
#else
  stage(0, 0);
#endif
  __syncthreads();

  for (int s = 0; s < kS; s += 32) {
    int cur = (s >> 5) & 1, nxt = cur ^ 1;
    bool more = (s + 32) < kS;
#if USE_TDM
    if (more && issuer) stage(nxt, s + 32);
#else
    if (more) stage(nxt, s + 32);
#endif
    // A = K^T fragment: a(mm=d2, k=s') = shK[cur][k*64 + td2*16 + mm]
    unsigned ua[8];
#pragma unroll
    for (int j = 0; j < 4; ++j) {
      int k0 = 16 * g + 2 * j;
      ua[j] = (unsigned)shK[cur][k0 * 64 + td2 * 16 + m] |
              ((unsigned)shK[cur][(k0 + 1) * 64 + td2 * 16 + m] << 16);
      int k1 = 16 * g + 8 + 2 * j;
      ua[j + 4] = (unsigned)shK[cur][k1 * 64 + td2 * 16 + m] |
                  ((unsigned)shK[cur][(k1 + 1) * 64 + td2 * 16 + m] << 16);
    }
    v8u va = {ua[0],ua[1],ua[2],ua[3],ua[4],ua[5],ua[6],ua[7]};
    v16bf af = __builtin_bit_cast(v16bf, va);
    // B = V fragments for the two column tiles (b(k,n) = shV[k*64 + td*16 + n])
#pragma unroll
    for (int t = 0; t < 2; ++t) {
      int td = tc0 + t;
      unsigned ub[8];
#pragma unroll
      for (int j = 0; j < 8; ++j) {
        int kk = 16 * g + 2 * j;
        ub[j] = (unsigned)shV[cur][kk * 64 + td * 16 + m] |
                ((unsigned)shV[cur][(kk + 1) * 64 + td * 16 + m] << 16);
      }
      v8u vb = {ub[0],ub[1],ub[2],ub[3],ub[4],ub[5],ub[6],ub[7]};
      if (t == 0) acc0 = WMMA_BF16(af, __builtin_bit_cast(v16bf, vb), acc0);
      else        acc1 = WMMA_BF16(af, __builtin_bit_cast(v16bf, vb), acc1);
    }
    if (more) {
#if USE_TDM
      if (issuer) __builtin_amdgcn_s_wait_tensorcnt(0);
#endif
      __syncthreads();
    }
  }

  unsigned short* out = ktv + (size_t)bh * kDH * kDH;
  int rb = td2 * 16 + (g << 3);
#pragma unroll
  for (int r = 0; r < 8; ++r) {
    out[(rb + r) * kDH + tc0 * 16 + m]       = f32_to_bf16(acc0[r] * 0.125f);
    out[(rb + r) * kDH + (tc0 + 1) * 16 + m] = f32_to_bf16(acc1[r] * 0.125f);
  }
}

// ---------------- imv[b, s, h*64+d] = Q[b,s,h,:] @ KtV[bh]  (KtV in LDS) ----------------
__global__ __launch_bounds__(256)
void imv_kernel(const unsigned short* __restrict__ qkv,
                const unsigned short* __restrict__ ktv,
                unsigned short* __restrict__ imv) {
  __shared__ __align__(16) unsigned short kt_sh[kDH * kDH];   // 8 KB
  int bh = blockIdx.y;
  int b = bh >> 4, h = bh & 15;
  {
    const unsigned int* src = (const unsigned int*)(ktv + (size_t)bh * kDH * kDH);
    unsigned int* dst = (unsigned int*)kt_sh;
    for (int i = threadIdx.x; i < (kDH * kDH / 2); i += blockDim.x) dst[i] = src[i];
  }
  __syncthreads();
  int wave = (int)threadIdx.x >> 5;
  int lane = (int)(threadIdx.x & 31u);
  int t  = blockIdx.x * (blockDim.x >> 5) + wave;  // 0..511 per head
  int ts = t >> 2, td = t & 3;
  const unsigned short* Qb = qkv + ((size_t)(b * kS + ts * 16)) * kN3 + h * kDH;
  v8f acc = {0.f, 0.f, 0.f, 0.f, 0.f, 0.f, 0.f, 0.f};
  for (int k = 0; k < kDH; k += 32) {
    v16bf af = load_a_frag(Qb + k, kN3);
    int n = lane & 15, g = lane >> 4;
    unsigned int u[8];
#pragma unroll
    for (int j = 0; j < 8; ++j) {
      int kk = k + 16 * g + 2 * j;
      u[j] = (unsigned int)kt_sh[kk * kDH + td * 16 + n] |
             ((unsigned int)kt_sh[(kk + 1) * kDH + td * 16 + n] << 16);
    }
    v8u ub = {u[0], u[1], u[2], u[3], u[4], u[5], u[6], u[7]};
    acc = WMMA_BF16(af, __builtin_bit_cast(v16bf, ub), acc);
  }
  int col   = h * kDH + td * 16 + (lane & 15);
  int rbase = b * kS + ts * 16 + ((lane >> 4) << 3);
#pragma unroll
  for (int r = 0; r < 8; ++r)
    imv[(size_t)(rbase + r) * kM + col] = f32_to_bf16(acc[r]);
}

// ---------------- host launcher ----------------
extern "C" void kernel_launch(void* const* d_in, const int* in_sizes, int n_in,
                              void* d_out, int out_size, void* d_ws, size_t ws_size,
                              hipStream_t stream) {
  (void)in_sizes; (void)n_in; (void)out_size; (void)ws_size;
  const float* savespace = (const float*)d_in[1];
  const float* Wqkv  = (const float*)d_in[2];
  const float* Wo    = (const float*)d_in[3];
  const float* ln1_g = (const float*)d_in[4];
  const float* ln1_b = (const float*)d_in[5];
  const float* ln2_g = (const float*)d_in[6];
  const float* ln2_b = (const float*)d_in[7];
  const float* fc1_w = (const float*)d_in[8];
  const float* fc1_b = (const float*)d_in[9];
  const float* fc2_w = (const float*)d_in[10];
  const float* fc2_b = (const float*)d_in[11];

  char* ws = (char*)d_ws;
  size_t off = 0;
  auto carve = [&](size_t bytes) -> void* {
    void* p = ws + off;
    off += (bytes + 255) & ~(size_t)255;
    return p;
  };
  unsigned short* h_bf    = (unsigned short*)carve((size_t)kRows * kM * 2);
  unsigned short* wqkv_bf = (unsigned short*)carve((size_t)kN3 * kM * 2);
  unsigned short* wo_bf   = (unsigned short*)carve((size_t)kM * kM * 2);
  unsigned short* fc1_bf  = (unsigned short*)carve((size_t)kHID * kM * 2);
  unsigned short* fc2_bf  = (unsigned short*)carve((size_t)kM * kHID * 2);
  unsigned short* qkv_bf  = (unsigned short*)carve((size_t)kRows * kN3 * 2);
  unsigned short* ktv_bf  = (unsigned short*)carve((size_t)kB * kH * kDH * kDH * 2);
  unsigned short* imv_bf  = (unsigned short*)carve((size_t)kRows * kM * 2);
  float*          ss_f    = (float*)carve((size_t)kRows * kM * 4);
  unsigned short* z_bf    = (unsigned short*)carve((size_t)kRows * kM * 2);
  unsigned short* hmid_bf = (unsigned short*)carve((size_t)kRows * kHID * 2);

  // 1) weights -> bf16
  cvt_f32_bf16<<<1024, 256, 0, stream>>>(Wqkv,  wqkv_bf, kN3 * kM);
  cvt_f32_bf16<<<1024, 256, 0, stream>>>(Wo,    wo_bf,   kM * kM);
  cvt_f32_bf16<<<1024, 256, 0, stream>>>(fc1_w, fc1_bf,  kHID * kM);
  cvt_f32_bf16<<<1024, 256, 0, stream>>>(fc2_w, fc2_bf,  kM * kHID);

  // 2) LN1 -> bf16
  ln_to_bf16<<<kRows, 256, 0, stream>>>(savespace, ln1_g, ln1_b, h_bf);

  // 3) qkv = h @ Wqkv^T   [8192 x 3072], K=1024
  gemm_xwT<0><<<dim3(kN3 / 64, kRows / 128), 256, 0, stream>>>(
      h_bf, wqkv_bf, kM, kN3, nullptr, nullptr, qkv_bf);

  // 4) KtV per head (scaled by 1/8): one block per (b,h)
  ktv_kernel<<<kB * kH, 256, 0, stream>>>(qkv_bf, ktv_bf);

  // 5) imv = Q @ KtV
  imv_kernel<<<dim3(64, kB * kH), 256, 0, stream>>>(qkv_bf, ktv_bf, imv_bf);

  // 6) ss = imv @ Wo^T + savespace  (fp32), K=1024
  gemm_xwT<1><<<dim3(kM / 64, kRows / 128), 256, 0, stream>>>(
      imv_bf, wo_bf, kM, kM, nullptr, savespace, ss_f);

  // 7) LN2 -> bf16
  ln_to_bf16<<<kRows, 256, 0, stream>>>(ss_f, ln2_g, ln2_b, z_bf);

  // 8) hmid = gelu(z @ fc1^T + b1) -> bf16, K=1024
  gemm_xwT<2><<<dim3(kHID / 64, kRows / 128), 256, 0, stream>>>(
      z_bf, fc1_bf, kM, kHID, fc1_b, nullptr, hmid_bf);

  // 9) out = hmid @ fc2^T + b2 + ss  (fp32), K=4096
  gemm_xwT<3><<<dim3(kM / 64, kRows / 128), 256, 0, stream>>>(
      hmid_bf, fc2_bf, kHID, kM, fc2_b, ss_f, d_out);
}
